// pi_GNN_17386027614710
// MI455X (gfx1250) — matrix operand order repaired and checked by reading, not compiled, not run on hardware
//
#include <hip/hip_runtime.h>
#include <hip/hip_bf16.h>

typedef __attribute__((ext_vector_type(16))) _Float16 v16h;
typedef __attribute__((ext_vector_type(8)))  float    v8f;

#define B_      512
#define NMAX_   256
#define HID_    64
#define HN_     64
#define INDIM_  18
#define SINK_ITERS_ 100

// A-fragment K index for 16x16x32 f16 WMMA (contiguous order, matches B): lane = hi*16+lo
__device__ __forceinline__ int kmap(int i, int hi) {
    return hi * 16 + i;
}

// ---------------------------------------------------------------------------
// Kernel 1: sf = relu(x[:,16:18]@fc1+b), ff = relu(x[:,0:16]@fc2+b),
//           scoresS = relu(w1 @ sf^T), scoresF = relu(w2 @ ff^T)   (per batch)
// ---------------------------------------------------------------------------
__global__ void score_kernel(const float* __restrict__ x,
                             const float* __restrict__ fc1_w, const float* __restrict__ fc1_b,
                             const float* __restrict__ fc2_w, const float* __restrict__ fc2_b,
                             const float* __restrict__ w1,    const float* __restrict__ w2,
                             float* __restrict__ scoresS, float* __restrict__ scoresF)
{
    extern __shared__ char smem[];
    _Float16* sf  = (_Float16*)smem;     // 256*64
    _Float16* ff  = sf  + 256 * 64;      // 256*64
    _Float16* w1h = ff  + 256 * 64;      // 64*64
    _Float16* w2h = w1h + 64 * 64;       // 64*64

    const int b = blockIdx.x;
    const int t = threadIdx.x;

    // per-row tiny MLPs (row n == thread t)
    {
        const float* xr = x + ((size_t)b * NMAX_ + t) * INDIM_;
        float f[16];
#pragma unroll
        for (int d = 0; d < 16; ++d) f[d] = xr[d];
        const float s0 = xr[16], s1 = xr[17];
        for (int c = 0; c < HID_; ++c) {
            float sv = fmaf(s0, fc1_w[c], fmaf(s1, fc1_w[64 + c], fc1_b[c]));
            sf[t * HID_ + c] = (_Float16)fmaxf(sv, 0.f);
            float fv = fc2_b[c];
#pragma unroll
            for (int d = 0; d < 16; ++d) fv = fmaf(f[d], fc2_w[d * HID_ + c], fv);
            ff[t * HID_ + c] = (_Float16)fmaxf(fv, 0.f);
        }
    }
    for (int idx = t; idx < 64 * 64; idx += 256) {
        w1h[idx] = (_Float16)w1[idx];
        w2h[idx] = (_Float16)w2[idx];
    }
    __syncthreads();

    const int lane = t & 31, wave = t >> 5;
    const int lo = lane & 15, hi = lane >> 4;

    // 2 matrices x (4 m-tiles x 16 n-tiles) = 128 wave-tiles over 8 waves
    for (int tile = wave; tile < 128; tile += 8) {
        const int sel = tile >> 6;
        const int tt  = tile & 63;
        const int m0  = (tt >> 4) * 16;
        const int n0  = (tt & 15) * 16;
        const _Float16* wsrc = sel ? w2h : w1h;
        const _Float16* fsrc = sel ? ff  : sf;
        float* out           = sel ? scoresF : scoresS;

        v8f acc = {};
#pragma unroll
        for (int kc = 0; kc < 2; ++kc) {
            const int k0 = kc * 32;
            v16h a, bf;
#pragma unroll
            for (int i = 0; i < 16; ++i)
                a[i] = wsrc[(m0 + lo) * HID_ + k0 + kmap(i, hi)];
#pragma unroll
            for (int i = 0; i < 16; ++i)
                bf[i] = fsrc[(n0 + lo) * HID_ + k0 + hi * 16 + i];
            acc = __builtin_amdgcn_wmma_f32_16x16x32_f16(false, a, false, bf,
                                                         (short)0, acc, false, false);
        }
#pragma unroll
        for (int r = 0; r < 8; ++r)
            out[((size_t)b * HN_ + m0 + hi * 8 + r) * NMAX_ + n0 + lo] = fmaxf(acc[r], 0.f);
    }
}

// ---------------------------------------------------------------------------
// Kernel 2: log-domain Sinkhorn (100 iters) on 65x257 couplings, entirely in
// LDS. Writes P += 0.5*exp(Z+u+v-norm) for the 64x256 block only.
// ---------------------------------------------------------------------------
__global__ void sinkhorn_kernel(const float* __restrict__ scores,
                                const float* __restrict__ bin_score,
                                float* __restrict__ P, int accumulate)
{
    extern __shared__ char smem[];
    float* Z = (float*)smem;        // 65*257
    float* u = Z + 65 * 257;        // 65
    float* v = u + 65;              // 257

    const int b = blockIdx.x;
    const int t = threadIdx.x;
    const float alpha    = bin_score[0];
    const float norm     = -logf(320.0f);          // -log(m+n), m=64 n=256
    const float lmu_last = logf(256.0f) + norm;
    const float lnu_last = logf(64.0f) + norm;

    for (int idx = t; idx < 65 * 257; idx += 256) {
        const int i = idx / 257, j = idx % 257;
        Z[idx] = (i < 64 && j < 256) ? scores[((size_t)b * HN_ + i) * NMAX_ + j] : alpha;
    }
    for (int j = t; j < 257; j += 256) v[j] = 0.f;
    __syncthreads();

    const int lane = t & 31, wave = t >> 5;

    for (int it = 0; it < SINK_ITERS_; ++it) {
        // u[i] = log_mu[i] - logsumexp_j(Z[i][j] + v[j])
        for (int i = wave; i < 65; i += 8) {
            float m = -3.0e38f, s = 0.f;
            const float* zr = Z + i * 257;
            for (int j = lane; j < 257; j += 32) {
                const float val = zr[j] + v[j];
                if (val > m) { s = s * __expf(m - val) + 1.f; m = val; }
                else         { s += __expf(val - m); }
            }
#pragma unroll
            for (int off = 16; off > 0; off >>= 1) {
                const float m2 = __shfl_xor(m, off, 32);
                const float s2 = __shfl_xor(s, off, 32);
                const float mm = fmaxf(m, m2);
                s = s * __expf(m - mm) + s2 * __expf(m2 - mm);
                m = mm;
            }
            if (lane == 0)
                u[i] = ((i < 64) ? norm : lmu_last) - (m + __logf(s));
        }
        __syncthreads();
        // v[j] = log_nu[j] - logsumexp_i(Z[i][j] + u[i])   (stride 257 % 64 == 1: no bank conflicts)
        for (int j = t; j < 257; j += 256) {
            float m = -3.0e38f, s = 0.f;
            for (int i = 0; i < 65; ++i) {
                const float val = Z[i * 257 + j] + u[i];
                if (val > m) { s = s * __expf(m - val) + 1.f; m = val; }
                else         { s += __expf(val - m); }
            }
            v[j] = ((j < 256) ? norm : lnu_last) - (m + __logf(s));
        }
        __syncthreads();
    }

    for (int idx = t; idx < HN_ * NMAX_; idx += 256) {
        const int i = idx >> 8, j = idx & 255;
        const float val = 0.5f * __expf(Z[i * 257 + j] + u[i] + v[j] - norm);
        const size_t o = (size_t)b * HN_ * NMAX_ + idx;
        P[o] = accumulate ? (P[o] + val) : val;
    }
}

// ---------------------------------------------------------------------------
// Kernel 3: adj_al = LN1((Pc@adj)@Pc^T), feats_al = LN2(Pc@x)   (per batch)
// ---------------------------------------------------------------------------
__global__ void align_kernel(const float* __restrict__ P, const float* __restrict__ adj,
                             const float* __restrict__ x,
                             const float* __restrict__ ln1_g, const float* __restrict__ ln1_b,
                             const float* __restrict__ ln2_g, const float* __restrict__ ln2_b,
                             float* __restrict__ Aln, float* __restrict__ Fln)
{
    extern __shared__ char smem[];
    _Float16* pch = (_Float16*)smem;         // 64*256  (Pc, f16)
    _Float16* t1h = pch + 64 * 256;          // 64*256  (Pc@adj, f16)
    _Float16* xh  = t1h + 64 * 256;          // 256*32  (x padded to 32 cols)
    float*    t2  = (float*)(xh + 256 * 32); // 64*64
    float*    fal = t2 + 64 * 64;            // 64*32
    float*    red = fal + 64 * 32;           // 32 scratch

    const int b = blockIdx.x;
    const int t = threadIdx.x;
    const int lane = t & 31, wave = t >> 5;
    const int lo = lane & 15, hi = lane >> 4;

    for (int idx = t; idx < 64 * 256; idx += 256)
        pch[idx] = (_Float16)P[(size_t)b * 64 * 256 + idx];
    for (int idx = t; idx < 256 * 32; idx += 256) {
        const int n = idx >> 5, d = idx & 31;
        xh[idx] = (d < INDIM_) ? (_Float16)x[((size_t)b * NMAX_ + n) * INDIM_ + d]
                               : (_Float16)0.f;
    }
    __syncthreads();

    // Phase 1: T1(64x256) = Pc @ adj ; adj f32 streamed from global (L2-resident)
    const float* adjb = adj + (size_t)b * NMAX_ * NMAX_;
    for (int tile = wave; tile < 64; tile += 8) {
        const int m0 = (tile >> 4) * 16, n0 = (tile & 15) * 16;
        v8f acc = {};
        for (int kc = 0; kc < 8; ++kc) {
            const int k0 = kc * 32;
            v16h a, bf;
#pragma unroll
            for (int i = 0; i < 16; ++i)
                a[i] = pch[(m0 + lo) * 256 + k0 + kmap(i, hi)];
#pragma unroll
            for (int i = 0; i < 16; ++i)
                bf[i] = (_Float16)adjb[(size_t)(k0 + hi * 16 + i) * 256 + n0 + lo];
            acc = __builtin_amdgcn_wmma_f32_16x16x32_f16(false, a, false, bf,
                                                         (short)0, acc, false, false);
        }
#pragma unroll
        for (int r = 0; r < 8; ++r)
            t1h[(m0 + hi * 8 + r) * 256 + n0 + lo] = (_Float16)acc[r];
    }
    __syncthreads();

    // Phase 2: T2(64x64) = T1 @ Pc^T  (B[k][l] = Pc[l][k])
    for (int tile = wave; tile < 16; tile += 8) {
        const int m0 = (tile >> 2) * 16, n0 = (tile & 3) * 16;
        v8f acc = {};
        for (int kc = 0; kc < 8; ++kc) {
            const int k0 = kc * 32;
            v16h a, bf;
#pragma unroll
            for (int i = 0; i < 16; ++i)
                a[i] = t1h[(m0 + lo) * 256 + k0 + kmap(i, hi)];
#pragma unroll
            for (int i = 0; i < 16; ++i)
                bf[i] = pch[(n0 + lo) * 256 + k0 + hi * 16 + i];
            acc = __builtin_amdgcn_wmma_f32_16x16x32_f16(false, a, false, bf,
                                                         (short)0, acc, false, false);
        }
#pragma unroll
        for (int r = 0; r < 8; ++r)
            t2[(m0 + hi * 8 + r) * 64 + n0 + lo] = acc[r];
    }

    // Phase 3: feats_al(64x18, padded 32) = Pc @ x
    {
        const int tile = wave; // 8 tiles, 8 waves
        const int m0 = (tile >> 1) * 16, n0 = (tile & 1) * 16;
        v8f acc = {};
        for (int kc = 0; kc < 8; ++kc) {
            const int k0 = kc * 32;
            v16h a, bf;
#pragma unroll
            for (int i = 0; i < 16; ++i)
                a[i] = pch[(m0 + lo) * 256 + k0 + kmap(i, hi)];
#pragma unroll
            for (int i = 0; i < 16; ++i)
                bf[i] = xh[(k0 + hi * 16 + i) * 32 + n0 + lo];
            acc = __builtin_amdgcn_wmma_f32_16x16x32_f16(false, a, false, bf,
                                                         (short)0, acc, false, false);
        }
#pragma unroll
        for (int r = 0; r < 8; ++r)
            fal[(m0 + hi * 8 + r) * 32 + n0 + lo] = acc[r];
    }
    __syncthreads();

    // LayerNorm 1 over t2[4096]
    {
        float ls = 0.f, lq = 0.f;
        for (int idx = t; idx < 4096; idx += 256) {
            const float vv = t2[idx];
            ls += vv; lq += vv * vv;
        }
#pragma unroll
        for (int off = 16; off > 0; off >>= 1) {
            ls += __shfl_xor(ls, off, 32);
            lq += __shfl_xor(lq, off, 32);
        }
        if (lane == 0) { red[wave] = ls; red[8 + wave] = lq; }
        __syncthreads();
        if (t == 0) {
            float S = 0.f, Q = 0.f;
            for (int w = 0; w < 8; ++w) { S += red[w]; Q += red[8 + w]; }
            const float mu  = S / 4096.f;
            const float var = Q / 4096.f - mu * mu;
            red[16] = mu;
            red[17] = rsqrtf(var + 1e-5f);
        }
        __syncthreads();
        const float mu = red[16], rs = red[17];
        for (int idx = t; idx < 4096; idx += 256)
            Aln[(size_t)b * 4096 + idx] = (t2[idx] - mu) * rs * ln1_g[idx] + ln1_b[idx];
        __syncthreads();
    }
    // LayerNorm 2 over fal (1152 valid elements)
    {
        float ls = 0.f, lq = 0.f;
        for (int idx = t; idx < 1152; idx += 256) {
            const float vv = fal[(idx / 18) * 32 + idx % 18];
            ls += vv; lq += vv * vv;
        }
#pragma unroll
        for (int off = 16; off > 0; off >>= 1) {
            ls += __shfl_xor(ls, off, 32);
            lq += __shfl_xor(lq, off, 32);
        }
        if (lane == 0) { red[wave] = ls; red[8 + wave] = lq; }
        __syncthreads();
        if (t == 0) {
            float S = 0.f, Q = 0.f;
            for (int w = 0; w < 8; ++w) { S += red[w]; Q += red[8 + w]; }
            const float mu  = S / 1152.f;
            const float var = Q / 1152.f - mu * mu;
            red[16] = mu;
            red[17] = rsqrtf(var + 1e-5f);
        }
        __syncthreads();
        const float mu = red[16], rs = red[17];
        for (int idx = t; idx < 1152; idx += 256)
            Fln[(size_t)b * 1152 + idx] =
                (fal[(idx / 18) * 32 + idx % 18] - mu) * rs * ln2_g[idx] + ln2_b[idx];
    }
}

// ---------------------------------------------------------------------------
// Generic WMMA GEMM: C(M,N) = act(A(M,lda) @ W(K,N) + bias). One 16x16 tile
// per wave; f32 operands converted to f16 fragments; K % 32 == 0.
// ---------------------------------------------------------------------------
__global__ void gemm_wmma(const float* __restrict__ A, int lda,
                          const float* __restrict__ W,
                          const float* __restrict__ bias,
                          float* __restrict__ C, int ldc,
                          int M, int N, int K, int do_relu)
{
    const int t = threadIdx.x, lane = t & 31, wave = t >> 5;
    const int lo = lane & 15, hi = lane >> 4;
    const int ntn   = (N + 15) >> 4;
    const int total = (M >> 4) * ntn;
    const int tile  = blockIdx.x * 8 + wave;
    if (tile >= total) return;
    const int m0  = (tile / ntn) * 16;
    const int n0  = (tile % ntn) * 16;
    const int col = n0 + lo;

    v8f acc = {};
    for (int k0 = 0; k0 < K; k0 += 32) {
        const float* arow = A + (size_t)(m0 + lo) * lda + k0;
        if (k0 + 32 < K) __builtin_prefetch(arow + 32, 0, 3); // global_prefetch_b8
        v16h a, bf;
#pragma unroll
        for (int i = 0; i < 16; ++i)
            a[i] = (_Float16)arow[kmap(i, hi)];
#pragma unroll
        for (int i = 0; i < 16; ++i)
            bf[i] = (col < N) ? (_Float16)W[(size_t)(k0 + hi * 16 + i) * N + col]
                              : (_Float16)0.f;
        acc = __builtin_amdgcn_wmma_f32_16x16x32_f16(false, a, false, bf,
                                                     (short)0, acc, false, false);
    }
    if (col < N) {
        const float bv = bias[col];
#pragma unroll
        for (int r = 0; r < 8; ++r) {
            float vv = acc[r] + bv;
            if (do_relu) vv = fmaxf(vv, 0.f);
            C[(size_t)(m0 + hi * 8 + r) * ldc + col] = vv;
        }
    }
}

// ---------------------------------------------------------------------------
// Kernel 5: row-wise log_softmax over 10 classes
// ---------------------------------------------------------------------------
__global__ void logsoftmax_kernel(const float* __restrict__ logits, float* __restrict__ out)
{
    const int row = blockIdx.x * blockDim.x + threadIdx.x;
    if (row >= B_) return;
    const float* z = logits + row * 10;
    float m = z[0];
#pragma unroll
    for (int j = 1; j < 10; ++j) m = fmaxf(m, z[j]);
    float s = 0.f;
#pragma unroll
    for (int j = 0; j < 10; ++j) s += __expf(z[j] - m);
    const float ls = m + __logf(s);
#pragma unroll
    for (int j = 0; j < 10; ++j) out[row * 10 + j] = z[j] - ls;
}

// ---------------------------------------------------------------------------
extern "C" void kernel_launch(void* const* d_in, const int* in_sizes, int n_in,
                              void* d_out, int out_size, void* d_ws, size_t ws_size,
                              hipStream_t stream) {
    const float* x     = (const float*)d_in[0];
    const float* adj   = (const float*)d_in[1];
    const float* fc1_w = (const float*)d_in[2];
    const float* fc1_b = (const float*)d_in[3];
    const float* fc2_w = (const float*)d_in[4];
    const float* fc2_b = (const float*)d_in[5];
    const float* w1    = (const float*)d_in[6];
    const float* w2    = (const float*)d_in[7];
    const float* bin   = (const float*)d_in[8];
    const float* ln1_g = (const float*)d_in[9];
    const float* ln1_b = (const float*)d_in[10];
    const float* fc3_w = (const float*)d_in[11];
    const float* fc3_b = (const float*)d_in[12];
    const float* fc4_w = (const float*)d_in[13];
    const float* fc4_b = (const float*)d_in[14];
    const float* ln2_g = (const float*)d_in[15];
    const float* ln2_b = (const float*)d_in[16];
    const float* fc5_w = (const float*)d_in[17];
    const float* fc5_b = (const float*)d_in[18];
    const float* fc6_w = (const float*)d_in[19];
    const float* fc6_b = (const float*)d_in[20];
    const float* fc7_w = (const float*)d_in[21];
    const float* fc7_b = (const float*)d_in[22];
    const float* fc8_w = (const float*)d_in[23];
    const float* fc8_b = (const float*)d_in[24];

    float* ws = (float*)d_ws;
    size_t off = 0;
    float* scoresS = ws + off; off += (size_t)B_ * HN_ * NMAX_;  // 8.39M f
    float* scoresF = ws + off; off += (size_t)B_ * HN_ * NMAX_;
    float* P       = ws + off; off += (size_t)B_ * HN_ * NMAX_;
    float* Aln     = ws + off; off += (size_t)B_ * 4096;
    float* Fln     = ws + off; off += (size_t)B_ * 1152;
    float* O1      = ws + off; off += (size_t)B_ * 256;
    float* O2      = ws + off; off += (size_t)B_ * 256;
    float* cat     = ws + off; off += (size_t)B_ * 256;
    float* O3      = ws + off; off += (size_t)B_ * 64;
    float* logits  = ws + off; off += (size_t)B_ * 10;

    const size_t lds1 = (size_t)(256*64 + 256*64 + 64*64 + 64*64) * sizeof(_Float16);
    score_kernel<<<B_, 256, lds1, stream>>>(x, fc1_w, fc1_b, fc2_w, fc2_b, w1, w2,
                                            scoresS, scoresF);

    const size_t lds2 = (size_t)(65*257 + 65 + 257) * sizeof(float);
    sinkhorn_kernel<<<B_, 256, lds2, stream>>>(scoresS, bin, P, 0);
    sinkhorn_kernel<<<B_, 256, lds2, stream>>>(scoresF, bin, P, 1);

    const size_t lds3 = (size_t)(64*256 + 64*256 + 256*32) * sizeof(_Float16)
                      + (size_t)(64*64 + 64*32 + 32) * sizeof(float);
    align_kernel<<<B_, 256, lds3, stream>>>(P, adj, x, ln1_g, ln1_b, ln2_g, ln2_b,
                                            Aln, Fln);

    auto gemm = [&](const float* A, int lda, const float* W, const float* bias,
                    float* C, int ldc, int M, int N, int K, int relu) {
        const int tiles = (M / 16) * ((N + 15) / 16);
        const int grid  = (tiles + 7) / 8;
        gemm_wmma<<<grid, 256, 0, stream>>>(A, lda, W, bias, C, ldc, M, N, K, relu);
    };
    gemm(Aln, 4096, fc3_w, fc3_b, O1,        256, B_, 256, 4096, 1);
    gemm(O1,  256,  fc4_w, fc4_b, cat,       256, B_, 128, 256,  1);
    gemm(Fln, 1152, fc5_w, fc5_b, O2,        256, B_, 256, 1152, 1);
    gemm(O2,  256,  fc6_w, fc6_b, cat + 128, 256, B_, 128, 256,  1);
    gemm(cat, 256,  fc7_w, fc7_b, O3,        64,  B_, 64,  256,  1);
    gemm(O3,  64,   fc8_w, fc8_b, logits,    10,  B_, 10,  64,   0);

    logsoftmax_kernel<<<2, 256, 0, stream>>>(logits, (float*)d_out);
}
